// LSTMModel_90220083019782
// MI455X (gfx1250) — matrix operand order repaired and checked by reading, not compiled, not run on hardware
//
#include <hip/hip_runtime.h>

typedef _Float16 half_t;
typedef __attribute__((ext_vector_type(16))) _Float16 v16h;
typedef __attribute__((ext_vector_type(8)))  _Float16 v8h;
typedef __attribute__((ext_vector_type(8)))  float    v8f;

#define V_   50000
#define E_   100
#define EPAD 128
#define H_   64
#define B_   256
#define T_   512
#define G_   256   // 4*H

// ---- CDNA5 async global->LDS path (guarded: falls back to prefetch) -------
#if defined(__has_builtin)
#  if __has_builtin(__builtin_amdgcn_global_load_async_to_lds_b128) && \
      __has_builtin(__builtin_amdgcn_s_wait_asynccnt)
#    define HAVE_ASYNC_LDS 1
#  endif
#endif
#ifndef HAVE_ASYNC_LDS
#  define HAVE_ASYNC_LDS 0
#endif

#if HAVE_ASYNC_LDS
typedef int v4i_ __attribute__((vector_size(4 * sizeof(int))));
typedef __attribute__((address_space(1))) v4i_ gv4i_t;   // global (device) side
typedef __attribute__((address_space(3))) v4i_ lv4i_t;   // LDS side

__device__ __forceinline__ void async_b128(const void* g, void* l) {
  // per-lane 16B: LDS[l] = MEM[g]; tracked by ASYNCcnt
  __builtin_amdgcn_global_load_async_to_lds_b128(
      (gv4i_t*)(unsigned long long)(size_t)g,
      (lv4i_t*)(unsigned int)(size_t)l,
      0, 0);
}
#endif

__device__ __forceinline__ float sigmoidf_(float x) { return 1.0f / (1.0f + __expf(-x)); }

// ---------------------------------------------------------------------------
// Prep: weights -> f16 [K][N] (B-matrix layout), pad E->128, pre-sum biases.
// ---------------------------------------------------------------------------
__global__ void prep_kernel(const float* w_ih0, const float* w_hh0,
                            const float* w_ih1, const float* w_hh1,
                            const float* b_ih0, const float* b_hh0,
                            const float* b_ih1, const float* b_hh1,
                            half_t* wih0h, half_t* whh0h, half_t* wih1h, half_t* whh1h,
                            float* b0, float* b1) {
  const int tid = blockIdx.x * blockDim.x + threadIdx.x;
  const int nth = gridDim.x * blockDim.x;
  for (int idx = tid; idx < EPAD * G_; idx += nth) {
    int k = idx / G_, n = idx % G_;
    wih0h[idx] = (k < E_) ? (half_t)w_ih0[n * E_ + k] : (half_t)0.0f;
  }
  for (int idx = tid; idx < H_ * G_; idx += nth) {
    int k = idx / G_, n = idx % G_;
    whh0h[idx] = (half_t)w_hh0[n * H_ + k];
    wih1h[idx] = (half_t)w_ih1[n * H_ + k];
    whh1h[idx] = (half_t)w_hh1[n * H_ + k];
  }
  for (int n = tid; n < G_; n += nth) {
    b0[n] = b_ih0[n] + b_hh0[n];
    b1[n] = b_ih1[n] + b_hh1[n];
  }
}

// ---------------------------------------------------------------------------
// Gate GEMM producing TIME-MAJOR xg[t][b][256] = A @ W + bias.
//   MODE 0: A row (b,t) = embedding gather of token x[b*T+t]  (K pad 100->128)
//   MODE 1: A row (b,t) = h1seq[t][b][0:64] (f16, K=64)
// Block = (t, 16-batch group); 16 waves, one 16x16 tile each; stores land in
// one contiguous 16KB slab.
// ---------------------------------------------------------------------------
template <int MODE>
__global__ __launch_bounds__(512)
void gate_gemm_kernel(const int* __restrict__ x, const float* __restrict__ emb,
                      const half_t* __restrict__ h1seq,
                      const half_t* __restrict__ wB, const float* __restrict__ bias,
                      float* __restrict__ xg) {
  __shared__ __attribute__((aligned(64))) half_t ldsA[16 * EPAD];
  const int KS  = (MODE == 0) ? EPAD : H_;
  const int tid = threadIdx.x;
  const int t   = blockIdx.x >> 4;
  const int b0  = (blockIdx.x & 15) * 16;

  if (MODE == 0) {
    for (int idx = tid; idx < 16 * EPAD; idx += 512) {
      int r = idx >> 7, k = idx & 127;
      int tok = x[(size_t)(b0 + r) * T_ + t];
      ldsA[idx] = (k < E_) ? (half_t)emb[(size_t)tok * E_ + k] : (half_t)0.0f;
    }
  } else {
    for (int idx = tid; idx < 16 * H_; idx += 512) {
      int r = idx >> 6, k = idx & 63;
      ldsA[r * H_ + k] = h1seq[((size_t)t * B_ + b0 + r) * H_ + k];
    }
  }
  __syncthreads();

  const int w    = tid >> 5, lane = tid & 31;
  const int n0   = w * 16;
  const int sel  = (lane & 16) ? 8 : 0;   // f16 A-matrix lane-half K select
  const int arow = lane & 15;
  const int rhi  = (lane & 16) ? 8 : 0;   // C/D row offset for lanes 16-31

  v8f acc = {};
  for (int kb = 0; kb < KS; kb += 32) {
    union { v16h v; v8h p[2]; } ua;
    ua.p[0] = *(const v8h*)(ldsA + arow * KS + kb + sel);
    ua.p[1] = *(const v8h*)(ldsA + arow * KS + kb + 16 + sel);
    v16h bf;
    for (int s = 0; s < 16; ++s) bf[s] = wB[(size_t)(kb + sel + (s & 7) + ((s & 8) ? 16 : 0)) * G_ + n0 + (lane & 15)];
    acc = __builtin_amdgcn_wmma_f32_16x16x32_f16(false, ua.v, false, bf,
                                                 (short)0, acc, false, false);
  }

  const int   col = n0 + (lane & 15);
  const float bv  = bias[col];
#pragma unroll
  for (int i = 0; i < 8; ++i)
    xg[((size_t)t * B_ + b0 + i + rhi) * G_ + col] = acc[i] + bv;
}

// ---------------------------------------------------------------------------
// LSTM scan. One block = 16 batch rows for all T steps (recurrence is
// batch-row independent -> no inter-block sync). Per step: 16 waves, wave w
// computes gate cols [16w,16w+16) with 2 WMMAs (K=64) from f16 h + w_hh in
// LDS. Cell state in registers. xg slab (16KB/step) is double-buffered into
// LDS via GLOBAL_LOAD_ASYNC_TO_LDS_B128 (2 steps in flight), retired with
// s_wait_asynccnt.
//   STORE_SEQ=1: emit h sequence as f16, time-major (layer 0)
//   STORE_SEQ=0: keep last h, fused ReLU+FC+sigmoid epilogue (layer 1)
// ---------------------------------------------------------------------------
template <int STORE_SEQ>
__global__ __launch_bounds__(512)
void lstm_scan_kernel(const float* __restrict__ xg, const half_t* __restrict__ whh,
                      half_t* __restrict__ hseq_out,
                      const float* __restrict__ fc_w, const float* __restrict__ fc_b,
                      float* __restrict__ out) {
  __shared__ __attribute__((aligned(64))) half_t ldsH[16 * H_];
  __shared__ __attribute__((aligned(64))) half_t ldsW[H_ * G_];
  __shared__ __attribute__((aligned(64))) float  ldsG[16 * G_];
#if HAVE_ASYNC_LDS
  __shared__ __attribute__((aligned(64))) float  ldsX[2][16 * G_];
#endif

  const int tid = threadIdx.x;
  const int rb  = blockIdx.x * 16;

  for (int idx = tid; idx < H_ * G_; idx += 512) ldsW[idx] = whh[idx];
  for (int idx = tid; idx < 16 * H_; idx += 512) ldsH[idx] = (half_t)0.0f;

  const int w    = tid >> 5, lane = tid & 31;
  const int n0   = w * 16;
  const int sel  = (lane & 16) ? 8 : 0;
  const int arow = lane & 15;
  const int rhi  = (lane & 16) ? 8 : 0;
  const int colc = n0 + (lane & 15);

  // per-lane persistent cell state: two adjacent elements of the 16x64 c
  const int e0 = tid * 2;
  const int r0 = e0 >> 6, c0 = e0 & 63, c1 = c0 + 1;
  float cc0 = 0.0f, cc1 = 0.0f;

#if HAVE_ASYNC_LDS
  auto issue_slab = [&](int t) {
    const float* g = xg + (size_t)t * (B_ * G_) + (size_t)rb * G_ + tid * 8;
    float*       l = &ldsX[t & 1][tid * 8];
    async_b128(g, l);            // 2 async b128 per thread = 16KB slab/block
    async_b128(g + 4, l + 4);
  };
  issue_slab(0);
  issue_slab(1);
#endif

  for (int t = 0; t < T_; ++t) {
#if HAVE_ASYNC_LDS
    if (t + 1 < T_) __builtin_amdgcn_s_wait_asynccnt(2);  // slab t done, t+1 in flight
    else            __builtin_amdgcn_s_wait_asynccnt(0);
#endif
    __syncthreads();   // slab t visible to all waves; h(t-1) writes visible

    v8f acc;
#if HAVE_ASYNC_LDS
    const float* slab = ldsX[t & 1];
#pragma unroll
    for (int i = 0; i < 8; ++i)
      acc[i] = slab[(i + rhi) * G_ + colc];
#else
    const float* slab = xg + (size_t)t * (B_ * G_) + (size_t)rb * G_;
#pragma unroll
    for (int i = 0; i < 8; ++i)
      acc[i] = slab[(i + rhi) * G_ + colc];
    if (t + 1 < T_)
      __builtin_prefetch(slab + B_ * G_ + tid * 8, 0, 0);
#endif

#pragma unroll
    for (int kb = 0; kb < H_; kb += 32) {
      union { v16h v; v8h p[2]; } ua;
      ua.p[0] = *(const v8h*)(ldsH + arow * H_ + kb + sel);
      ua.p[1] = *(const v8h*)(ldsH + arow * H_ + kb + 16 + sel);
      v16h bf;
      for (int s = 0; s < 16; ++s) bf[s] = ldsW[(kb + sel + (s & 7) + ((s & 8) ? 16 : 0)) * G_ + n0 + (lane & 15)];
      acc = __builtin_amdgcn_wmma_f32_16x16x32_f16(false, ua.v, false, bf,
                                                   (short)0, acc, false, false);
    }
#pragma unroll
    for (int i = 0; i < 8; ++i)
      ldsG[(i + rhi) * G_ + colc] = acc[i];
    __syncthreads();   // gates ready; all slab-t reads retired

#if HAVE_ASYNC_LDS
    if (t + 2 < T_) issue_slab(t + 2);   // refill the buffer just freed
#endif

    {
      float ig = sigmoidf_(ldsG[r0 * G_ + c0]);
      float fg = sigmoidf_(ldsG[r0 * G_ + 64 + c0]);
      float gg = tanhf(ldsG[r0 * G_ + 128 + c0]);
      float og = sigmoidf_(ldsG[r0 * G_ + 192 + c0]);
      cc0 = fg * cc0 + ig * gg;
      float hv = og * tanhf(cc0);
      ldsH[r0 * H_ + c0] = (half_t)hv;
      if (STORE_SEQ)        hseq_out[((size_t)t * B_ + rb + r0) * H_ + c0] = (half_t)hv;
      else if (t == T_ - 1) ldsG[r0 * G_ + c0] = hv;   // stash last h in consumed i-slot

      float ig1 = sigmoidf_(ldsG[r0 * G_ + c1]);
      float fg1 = sigmoidf_(ldsG[r0 * G_ + 64 + c1]);
      float gg1 = tanhf(ldsG[r0 * G_ + 128 + c1]);
      float og1 = sigmoidf_(ldsG[r0 * G_ + 192 + c1]);
      cc1 = fg1 * cc1 + ig1 * gg1;
      float hv1 = og1 * tanhf(cc1);
      ldsH[r0 * H_ + c1] = (half_t)hv1;
      if (STORE_SEQ)        hseq_out[((size_t)t * B_ + rb + r0) * H_ + c1] = (half_t)hv1;
      else if (t == T_ - 1) ldsG[r0 * G_ + c1] = hv1;
    }
  }
  __syncthreads();

  if (!STORE_SEQ && tid < 16) {
    float a = fc_b[0];
    for (int k = 0; k < H_; ++k) {
      float hv = ldsG[tid * G_ + k];
      hv = hv > 0.0f ? hv : 0.0f;
      a += hv * fc_w[k];
    }
    out[rb + tid] = 1.0f / (1.0f + __expf(-a));
  }
}

// ---------------------------------------------------------------------------
extern "C" void kernel_launch(void* const* d_in, const int* in_sizes, int n_in,
                              void* d_out, int out_size, void* d_ws, size_t ws_size,
                              hipStream_t stream) {
  const int*   x     = (const int*)  d_in[0];
  const float* emb   = (const float*)d_in[1];
  const float* w_ih0 = (const float*)d_in[2];
  const float* w_hh0 = (const float*)d_in[3];
  const float* b_ih0 = (const float*)d_in[4];
  const float* b_hh0 = (const float*)d_in[5];
  const float* w_ih1 = (const float*)d_in[6];
  const float* w_hh1 = (const float*)d_in[7];
  const float* b_ih1 = (const float*)d_in[8];
  const float* b_hh1 = (const float*)d_in[9];
  const float* fc_w  = (const float*)d_in[10];
  const float* fc_b  = (const float*)d_in[11];
  float* out = (float*)d_out;

  char*  ws  = (char*)d_ws;
  size_t off = 0;
  auto alloc = [&](size_t bytes) -> char* {
    char* p = ws + off;
    off += (bytes + 255) & ~(size_t)255;
    return p;
  };
  half_t* wih0h = (half_t*)alloc((size_t)EPAD * G_ * 2);
  half_t* whh0h = (half_t*)alloc((size_t)H_ * G_ * 2);
  half_t* wih1h = (half_t*)alloc((size_t)H_ * G_ * 2);
  half_t* whh1h = (half_t*)alloc((size_t)H_ * G_ * 2);
  float*  b0    = (float*) alloc((size_t)G_ * 4);
  float*  b1    = (float*) alloc((size_t)G_ * 4);
  float*  xg    = (float*) alloc((size_t)B_ * T_ * G_ * 4);   // time-major [T][B][G]
  half_t* h1seq = (half_t*)alloc((size_t)B_ * T_ * H_ * 2);   // time-major [T][B][H]

  prep_kernel<<<64, 256, 0, stream>>>(w_ih0, w_hh0, w_ih1, w_hh1,
                                      b_ih0, b_hh0, b_ih1, b_hh1,
                                      wih0h, whh0h, wih1h, whh1h, b0, b1);

  gate_gemm_kernel<0><<<T_ * 16, 512, 0, stream>>>(x, emb, nullptr, wih0h, b0, xg);
  lstm_scan_kernel<1><<<B_ / 16, 512, 0, stream>>>(xg, whh0h, h1seq, nullptr, nullptr, nullptr);
  gate_gemm_kernel<1><<<T_ * 16, 512, 0, stream>>>(nullptr, nullptr, h1seq, wih1h, b1, xg);
  lstm_scan_kernel<0><<<B_ / 16, 512, 0, stream>>>(xg, whh1h, nullptr, fc_w, fc_b, out);
}
